// Patch_Weather_Attention_36180804501600
// MI455X (gfx1250) — matrix operand order repaired and checked by reading, not compiled, not run on hardware
//
#include <hip/hip_runtime.h>

typedef __attribute__((ext_vector_type(16))) _Float16 v16h;
typedef __attribute__((ext_vector_type(8)))  float    v8f;

#define QD   24
#define KD   24
#define HD   64
#define NCTX 512
#define DOUT 128

static __device__ __forceinline__ float fast_tanh(float x) {
    // tanh(x) = (e^{2x}-1)/(e^{2x}+1), fast rcp instead of IEEE divide
    float e = __expf(2.0f * x);
    return (e - 1.0f) * __builtin_amdgcn_rcpf(e + 1.0f);
}

__global__ __launch_bounds__(256) void pwa_fused(
    const float* __restrict__ fut,   // [B,24]
    const float* __restrict__ past,  // [B,512,24]
    const float* __restrict__ tout,  // [B,512,128]
    const float* __restrict__ wq1, const float* __restrict__ bq1,
    const float* __restrict__ wq2, const float* __restrict__ bq2,
    const float* __restrict__ wk1, const float* __restrict__ bk1,
    const float* __restrict__ wk2, const float* __restrict__ bk2,
    float* __restrict__ ctx_out,     // [B,128]
    float* __restrict__ attn_out)    // [B,512]
{
    const int b    = blockIdx.x;
    const int tid  = threadIdx.x;
    const int lane = tid & 31;
    const int wave = tid >> 5;
    const int nl   = lane & 15;            // N index inside a 16-wide tile
    const int kb   = (lane < 16) ? 0 : 16; // B-layout K base
    const int row  = lane & 15;            // A-layout row inside a 16-row tile

    __shared__ float    s_hq[HD];
    __shared__ float    s_q[HD];
    __shared__ float    s_score[NCTX];
    __shared__ float    s_part[8];
    __shared__ float    s_bcast;
    __shared__ _Float16 s_a[8][16 * 32];    // per-wave A staging (16 rows x 32, cols 24..31 zero)
    __shared__ _Float16 s_hid[8][16 * HD];  // per-wave hidden restage (f16)
    __shared__ float4   s_cacc[256];

    const float* tb = tout + (size_t)b * NCTX * DOUT;

    // Early L2 prefetch of this batch's transformer_output slice (256 KB),
    // overlapped with the WMMA MLP below.
    for (int n = tid; n < NCTX; n += 256)
        __builtin_prefetch(tb + (size_t)n * DOUT, 0, 1);

    // Zero the K-pad columns (24..31) of the per-wave A staging buffer once.
    {
        const int zr = lane >> 1;
        const int zc = 24 + (lane & 1) * 4;
        #pragma unroll
        for (int j = 0; j < 4; ++j)
            s_a[wave][zr * 32 + zc + j] = (_Float16)0.0f;
    }

    // ---------------- Q path: q = tanh(fut@wq1+bq1)@wq2+bq2 ----------------
    if (tid < HD) {
        float a = bq1[tid];
        #pragma unroll
        for (int i = 0; i < QD; ++i)
            a = fmaf(fut[(size_t)b * QD + i], wq1[i * HD + tid], a);
        s_hq[tid] = fast_tanh(a);
    }
    __syncthreads();
    if (tid < HD) {
        float a = bq2[tid];
        #pragma unroll
        for (int j = 0; j < HD; ++j)
            a = fmaf(s_hq[j], wq2[j * HD + tid], a);
        s_q[tid] = a;
    }
    __syncthreads();

    // ---------------- Load B matrices (weights) into registers -------------
    // All global loads use always-valid addresses; padding handled by value
    // selects (v_cndmask), never by exec-masked loads.
    // GEMM1 B: wk1 zero-padded 32x64 -> 4 column tiles of 32x16 f16
    v16h B1[4];
    #pragma unroll
    for (int t = 0; t < 4; ++t) {
        #pragma unroll
        for (int e = 0; e < 16; ++e) {
            const int k  = kb + e;
            const int kc = (k < KD) ? k : 0;               // clamped, always valid
            const float v = wk1[kc * HD + t * 16 + nl];
            B1[t][e] = (k < KD) ? (_Float16)v : (_Float16)0.0f;
        }
    }
    // GEMM2 B: wk2 64x64 -> 4 column tiles x 2 K-chunks of 32x16 f16
    v16h B2[4][2];
    #pragma unroll
    for (int t = 0; t < 4; ++t) {
        #pragma unroll
        for (int c = 0; c < 2; ++c) {
            #pragma unroll
            for (int e = 0; e < 16; ++e) {
                const int k = c * 32 + kb + e;
                B2[t][c][e] = (_Float16)wk2[k * HD + t * 16 + nl];
            }
        }
    }

    // ---------------- K path MLP via WMMA, 4 row-tiles per wave -------------
    #pragma unroll 1
    for (int it = 0; it < 4; ++it) {
        const int tile = wave + it * 8;        // 0..31 (32 tiles of 16 rows)

        // Stage past[b, tile*16 .. +15, 0..23] (384 contiguous floats) into
        // the per-wave LDS A buffer: coalesced float4 loads, f16 stores.
        {
            const float* src = past + ((size_t)b * NCTX + (size_t)tile * 16) * KD;
            #pragma unroll
            for (int i = 0; i < 3; ++i) {
                const int f = (i * 32 + lane) * 4;         // 0..380, multiple of 4
                const float4 v = *(const float4*)(src + f);
                const int r = f / KD;                      // float4 never crosses a row
                const int c = f % KD;
                _Float16* dst = &s_a[wave][r * 32 + c];
                dst[0] = (_Float16)v.x; dst[1] = (_Float16)v.y;
                dst[2] = (_Float16)v.z; dst[3] = (_Float16)v.w;
            }
        }
        __builtin_amdgcn_wave_barrier();   // same-wave LDS is in-order; fence compiler

        // A (16x32 f16, K padded 24->32): unconditional LDS gather per layout
        v16h A;
        #pragma unroll
        for (int e = 0; e < 16; ++e) {
            const int k = e + ((lane < 16) ? 0 : 16);
            A[e] = s_a[wave][row * 32 + k];
        }

        // GEMM1: hidden = A @ wk1  (4 N-tiles)
        v8f h[4];
        #pragma unroll
        for (int t = 0; t < 4; ++t) {
            v8f z = {};
            h[t] = __builtin_amdgcn_wmma_f32_16x16x32_f16(
                false, A, false, B1[t], (short)0, z, false, false);
        }

        // bias + tanh, restage into LDS as f16 (D layout -> linear)
        #pragma unroll
        for (int t = 0; t < 4; ++t) {
            const float bv = bk1[t * 16 + nl];
            const int moff = (lane < 16) ? 0 : 8;
            #pragma unroll
            for (int r = 0; r < 8; ++r) {
                float hv = fast_tanh(h[t][r] + bv);
                s_hid[wave][(moff + r) * HD + t * 16 + nl] = (_Float16)hv;
            }
        }
        __builtin_amdgcn_wave_barrier();

        // Reload hidden as two 16x32 A matrices (K=0..31, K=32..63)
        v16h A2a, A2b;
        #pragma unroll
        for (int e = 0; e < 16; ++e) {
            const int k = e + ((lane < 16) ? 0 : 16);
            A2a[e] = s_hid[wave][row * HD + k];
            A2b[e] = s_hid[wave][row * HD + 32 + k];
        }

        // GEMM2: kfeat = hidden @ wk2 (+bias)
        v8f k2[4];
        #pragma unroll
        for (int t = 0; t < 4; ++t) {
            v8f z = {};
            v8f acc = __builtin_amdgcn_wmma_f32_16x16x32_f16(
                false, A2a, false, B2[t][0], (short)0, z, false, false);
            acc = __builtin_amdgcn_wmma_f32_16x16x32_f16(
                false, A2b, false, B2[t][1], (short)0, acc, false, false);
            const float bv = bk2[t * 16 + nl];
            #pragma unroll
            for (int r = 0; r < 8; ++r) k2[t][r] = acc[r] + bv;
        }

        // score[m] = sum_h kfeat[m,h] * q[h]; lane holds h = t*16+nl,
        // rows m = r (+8 for upper lanes). Reduce across 16-lane halves.
        float part[8];
        #pragma unroll
        for (int r = 0; r < 8; ++r) part[r] = 0.0f;
        #pragma unroll
        for (int t = 0; t < 4; ++t) {
            const float qv = s_q[t * 16 + nl];
            #pragma unroll
            for (int r = 0; r < 8; ++r) part[r] = fmaf(k2[t][r], qv, part[r]);
        }
        #pragma unroll
        for (int m = 1; m < 16; m <<= 1) {
            #pragma unroll
            for (int r = 0; r < 8; ++r)
                part[r] += __shfl_xor(part[r], m, 32);
        }
        if (nl == 0) {   // lanes 0 (rows 0..7) and 16 (rows 8..15)
            const int mbase = tile * 16 + ((lane < 16) ? 0 : 8);
            #pragma unroll
            for (int r = 0; r < 8; ++r)
                s_score[mbase + r] = part[r] * 0.125f;   // 1/sqrt(64)
        }
        __builtin_amdgcn_wave_barrier();   // next iter's LDS stores stay behind reads
    }
    __syncthreads();

    // ---------------- softmax over 512 scores ------------------------------
    float lmax = -3.4e38f;
    for (int n = tid; n < NCTX; n += 256) lmax = fmaxf(lmax, s_score[n]);
    #pragma unroll
    for (int m = 16; m >= 1; m >>= 1) lmax = fmaxf(lmax, __shfl_xor(lmax, m, 32));
    if (lane == 0) s_part[wave] = lmax;
    __syncthreads();
    if (tid == 0) {
        float v = s_part[0];
        #pragma unroll
        for (int i = 1; i < 8; ++i) v = fmaxf(v, s_part[i]);
        s_bcast = v;
    }
    __syncthreads();
    const float mx = s_bcast;

    float lsum = 0.0f;
    for (int n = tid; n < NCTX; n += 256) {
        float e = __expf(s_score[n] - mx);
        s_score[n] = e;
        lsum += e;
    }
    #pragma unroll
    for (int m = 16; m >= 1; m >>= 1) lsum += __shfl_xor(lsum, m, 32);
    __syncthreads();
    if (lane == 0) s_part[wave] = lsum;
    __syncthreads();
    if (tid == 0) {
        float v = 0.0f;
        #pragma unroll
        for (int i = 0; i < 8; ++i) v += s_part[i];
        s_bcast = 1.0f / v;
    }
    __syncthreads();
    const float inv = s_bcast;
    for (int n = tid; n < NCTX; n += 256) {
        float w = s_score[n] * inv;
        s_score[n] = w;
        attn_out[(size_t)b * NCTX + n] = w;
    }
    __syncthreads();

    // ---------------- context = attn @ transformer_output -------------------
    const int cg = tid & 31;   // float4 column group (32*4 = 128 cols)
    const int rg = tid >> 5;   // row group (8)
    float4 acc = make_float4(0.f, 0.f, 0.f, 0.f);
    for (int n = rg; n < NCTX; n += 8) {
        const float  w = s_score[n];
        const float4 v = ((const float4*)(tb + (size_t)n * DOUT))[cg];
        acc.x = fmaf(w, v.x, acc.x);
        acc.y = fmaf(w, v.y, acc.y);
        acc.z = fmaf(w, v.z, acc.z);
        acc.w = fmaf(w, v.w, acc.w);
    }
    s_cacc[tid] = acc;
    __syncthreads();
    if (tid < 32) {
        float4 t = s_cacc[tid];
        #pragma unroll
        for (int g = 1; g < 8; ++g) {
            float4 u = s_cacc[g * 32 + tid];
            t.x += u.x; t.y += u.y; t.z += u.z; t.w += u.w;
        }
        ((float4*)(ctx_out + (size_t)b * DOUT))[tid] = t;
    }
}

extern "C" void kernel_launch(void* const* d_in, const int* in_sizes, int n_in,
                              void* d_out, int out_size, void* d_ws, size_t ws_size,
                              hipStream_t stream) {
    const float* fut  = (const float*)d_in[0];
    const float* past = (const float*)d_in[1];
    const float* tout = (const float*)d_in[2];
    const float* wq1  = (const float*)d_in[3];
    const float* bq1  = (const float*)d_in[4];
    const float* wq2  = (const float*)d_in[5];
    const float* bq2  = (const float*)d_in[6];
    const float* wk1  = (const float*)d_in[7];
    const float* bk1  = (const float*)d_in[8];
    const float* wk2  = (const float*)d_in[9];
    const float* bk2  = (const float*)d_in[10];

    const int B = in_sizes[0] / QD;
    float* ctx  = (float*)d_out;                       // [B,128] first
    float* attn = ctx + (size_t)B * DOUT;              // then [B,1,512]

    pwa_fused<<<B, 256, 0, stream>>>(fut, past, tout,
                                     wq1, bq1, wq2, bq2,
                                     wk1, bk1, wk2, bk2,
                                     ctx, attn);
}
